// SeqEncDec_5446018531636
// MI455X (gfx1250) — matrix operand order repaired and checked by reading, not compiled, not run on hardware
//
#include <hip/hip_runtime.h>

// ---------------------------------------------------------------------------
// Seq2seq LSTM (enc 15 + dec 15 steps), H=128, IN=4, BATCH=32768.
// Each wave32 owns a 16-sample tile for the whole recurrence.
// Weights: one f16 LDS image (async-copied via GLOBAL_LOAD_ASYNC_TO_LDS_B128).
// Gate bias folded into Wx column 4 (x fragment carries a constant 1.0), so
// WMMA accumulators start from the inline-constant 0.
// c state: VGPRs in WMMA C/D layout. h: per-wave 16x128 f16 LDS buffer that
// doubles as the A-fragment source. Gates: v_wmma_f32_16x16x32_f16.
// ---------------------------------------------------------------------------

typedef __attribute__((ext_vector_type(16))) _Float16 v16h;
typedef __attribute__((ext_vector_type(8)))  float    v8f;

#define DI __device__ __forceinline__

constexpr int HDIM  = 128;
constexpr int G4    = 512;       // 4*H
constexpr int OBS   = 15;
constexpr int PRED  = 15;
constexpr int BATCH = 32768;
constexpr int WAVES = 8;         // waves per block
constexpr int TPB   = WAVES * 32;

// ---- workspace / LDS image layout (bytes), identical in d_ws and in LDS ----
constexpr size_t WHH_E_OFF = 0;                              // 512x128 f16
constexpr size_t WHH_D_OFF = WHH_E_OFF + (size_t)G4*HDIM*2;  // 512x128 f16
constexpr size_t WX_E_OFF  = WHH_D_OFF + (size_t)G4*HDIM*2;  // 512x8 f16: k0..3=Wih, k4=bias, k5..7=0
constexpr size_t WX_D_OFF  = WX_E_OFF  + (size_t)G4*8*2;
constexpr size_t OUTW_OFF  = WX_D_OFF  + (size_t)G4*8*2;     // 4x128 f16
constexpr size_t OUTB_OFF  = OUTW_OFF  + (size_t)4*HDIM*2;   // 16 f32 (out_b padded)
constexpr size_t COPY_BYTES= OUTB_OFF  + 16*4;               // = 279616
constexpr size_t HBUF_OFF  = COPY_BYTES;                     // WAVES * 16*128 f16
constexpr size_t XBUF_OFF  = HBUF_OFF + (size_t)WAVES*16*HDIM*2;
constexpr size_t SMEM_BYTES= XBUF_OFF + (size_t)WAVES*16*4*2;   // ~306 KB <= 320 KB

static_assert(SMEM_BYTES <= 320*1024, "LDS budget exceeded");
static_assert(COPY_BYTES % 16 == 0, "copy image must be 16B aligned");

// ---------------------------------------------------------------------------
// prep kernel: f32 -> f16 weight packing + bias fusion into d_ws (LDS image)
// ---------------------------------------------------------------------------
__global__ void seqlstm_prep(const float* __restrict__ eWih, const float* __restrict__ eWhh,
                             const float* __restrict__ ebih, const float* __restrict__ ebhh,
                             const float* __restrict__ dWih, const float* __restrict__ dWhh,
                             const float* __restrict__ dbih, const float* __restrict__ dbhh,
                             const float* __restrict__ oW,   const float* __restrict__ ob,
                             unsigned char* __restrict__ ws)
{
    int idx = blockIdx.x * blockDim.x + threadIdx.x;       // 0 .. 65535
    _Float16* whhE = (_Float16*)(ws + WHH_E_OFF);
    _Float16* whhD = (_Float16*)(ws + WHH_D_OFF);
    _Float16* wxE  = (_Float16*)(ws + WX_E_OFF);
    _Float16* wxD  = (_Float16*)(ws + WX_D_OFF);
    _Float16* outw = (_Float16*)(ws + OUTW_OFF);
    float*    obp  = (float*)(ws + OUTB_OFF);

    if (idx < G4*HDIM) {
        whhE[idx] = (_Float16)eWhh[idx];
        whhD[idx] = (_Float16)dWhh[idx];
    }
    if (idx < G4*8) {
        int n = idx >> 3, k = idx & 7;
        float ve = 0.0f, vd = 0.0f;
        if (k < 4)       { ve = eWih[n*4 + k];      vd = dWih[n*4 + k]; }
        else if (k == 4) { ve = ebih[n] + ebhh[n];  vd = dbih[n] + dbhh[n]; }
        wxE[idx] = (_Float16)ve;
        wxD[idx] = (_Float16)vd;
    }
    if (idx < 4*HDIM) outw[idx] = (_Float16)oW[idx];
    if (idx < 16) obp[idx] = (idx < 4) ? ob[idx] : 0.0f;
}

// ---------------------------------------------------------------------------
// helpers
// ---------------------------------------------------------------------------
DI float frcp(float x)  { return __builtin_amdgcn_rcpf(x); }        // v_rcp_f32
DI float fsig(float v)  { return frcp(1.0f + __expf(-v)); }
DI float ftanh(float v) { float e = __expf(-2.0f * v);              // (1-e)/(1+e)
                          return (1.0f - e) * frcp(1.0f + e); }

// 32 contiguous bytes -> one f16 B fragment (16 halves along K of row n)
DI v16h ld32h(const _Float16* p) {
    v16h r; __builtin_memcpy(&r, p, 32); return r;
}
// two separated 16B chunks -> A fragment (halves 0..7 then 8..15)
DI v16h ld2x16h(const _Float16* p0, const _Float16* p1) {
    v16h r;
    __builtin_memcpy(&r, p0, 16);
    __builtin_memcpy(((char*)&r) + 16, p1, 16);
    return r;
}
// A fragment of h from row-major 16x128 f16 LDS buffer
DI v16h load_hfrag(const _Float16* hb, int kf, int n16, int hi) {
    int kb = kf * 32 + hi * 16;
    const _Float16* row = hb + n16 * HDIM;
    return ld2x16h(row + kb, row + kb + 8);
}

// Async bulk copy global -> LDS (GLOBAL_LOAD_ASYNC_TO_LDS_B128, ASYNCcnt).
// VGLOBAL GV mode (ISA 15.18): vdst = LDS byte address VGPR, vaddr = 64b
// global address pair, SADDR = NULL ("off").
DI void async_g2l_b128(const unsigned char* g, unsigned char* l) {
    unsigned           lds = (unsigned)(unsigned long long)l;  // addr[31:0] = LDS addr
    unsigned long long ga  = (unsigned long long)g;
    asm volatile("global_load_async_to_lds_b128 %0, %1, off"
                 :: "v"(lds), "v"(ga) : "memory");
}
DI void wait_asynccnt0() {
#if __has_builtin(__builtin_amdgcn_s_wait_asynccnt)
    __builtin_amdgcn_s_wait_asynccnt(0);
#else
    asm volatile("s_wait_asynccnt 0" ::: "memory");
#endif
}

// One LSTM cell step for a 16-sample tile.
// c[8] is the cell state in C/D layout (tile jt covers hidden j = jt*16..+15).
// If h_to_c, c is overwritten with h_new (enc->dec handoff: c0 = h_enc).
DI void lstm_step(const _Float16* __restrict__ Whh, const _Float16* __restrict__ Wx,
                  _Float16* __restrict__ hb,
                  v16h ax, const v16h hf[4], v8f c[8],
                  int n16, int hi, bool h_to_c)
{
    const int koff = hi * 16;
    #pragma unroll
    for (int jt = 0; jt < 8; ++jt) {
        const int j0 = jt * 16;
        v8f acc[4];
        #pragma unroll
        for (int g = 0; g < 4; ++g) {
            const int n0 = g * HDIM + j0;            // row in the 512-wide gate block
            // x + bias contribution (K=5 padded to 32; lanes 0-15 carry data);
            // accumulator starts at inline-constant 0.
            v16h bx{};
            if (hi == 0) __builtin_memcpy(&bx, Wx + (size_t)(n0 + n16) * 8, 16);
            v8f a{};
            a = __builtin_amdgcn_wmma_f32_16x16x32_f16(false, ax, false, bx, (short)0, a, false, false);
            // h contribution, K = 128: preload all 4 B fragments, then chain WMMAs
            v16h b[4];
            #pragma unroll
            for (int kf = 0; kf < 4; ++kf)
                b[kf] = ld32h(Whh + (size_t)(n0 + n16) * HDIM + kf * 32 + koff);
            #pragma unroll
            for (int kf = 0; kf < 4; ++kf)
                a = __builtin_amdgcn_wmma_f32_16x16x32_f16(false, hf[kf], false, b[kf], (short)0, a, false, false);
            acc[g] = a;
        }
        // elementwise gate math in C/D layout (no shuffles needed)
        #pragma unroll
        for (int e = 0; e < 8; ++e) {
            float iv = fsig(acc[0][e]);
            float fv = fsig(acc[1][e]);
            float gv = ftanh(acc[2][e]);
            float ov = fsig(acc[3][e]);
            float cn = fv * c[jt][e] + iv * gv;
            float hn = ov * ftanh(cn);
            c[jt][e] = h_to_c ? hn : cn;
            // h_new -> row-major f16 LDS (row m = e + 8*hi, col j0+n16)
            hb[(e + 8 * hi) * HDIM + j0 + n16] = (_Float16)hn;
        }
    }
}

// ---------------------------------------------------------------------------
// main kernel: whole enc+dec recurrence, one 16-batch tile per wave
// ---------------------------------------------------------------------------
__global__ void __launch_bounds__(TPB)
seqlstm_main(const float* __restrict__ xg, const unsigned char* __restrict__ ws,
             float* __restrict__ out)
{
    extern __shared__ unsigned char smem[];
    const int tid  = threadIdx.x;
    const int lane = tid & 31;
    const int wv   = tid >> 5;
    const int n16  = lane & 15;
    const int hi   = lane >> 4;

    // ---- stage the weight image into LDS via async global->LDS DMA --------
    constexpr int CHUNKS = (int)(COPY_BYTES / 16);
    for (int i = tid; i < CHUNKS; i += TPB)
        async_g2l_b128(ws + (size_t)i * 16, smem + (size_t)i * 16);
    wait_asynccnt0();
    __syncthreads();

    const _Float16* WhhE = (const _Float16*)(smem + WHH_E_OFF);
    const _Float16* WhhD = (const _Float16*)(smem + WHH_D_OFF);
    const _Float16* WxE  = (const _Float16*)(smem + WX_E_OFF);
    const _Float16* WxD  = (const _Float16*)(smem + WX_D_OFF);
    const _Float16* outW = (const _Float16*)(smem + OUTW_OFF);
    const float*    obp  = (const float*)(smem + OUTB_OFF);
    _Float16* hb = (_Float16*)(smem + HBUF_OFF) + (size_t)wv * 16 * HDIM;
    _Float16* xb = (_Float16*)(smem + XBUF_OFF) + (size_t)wv * 16 * 4;

    const int tile = blockIdx.x * WAVES + wv;   // 16-sample batch tile
    const int base = tile * 16;

    // ---- init recurrent state: h = 0 (LDS), c = 0 (VGPRs)
    {
        uint4 z = {0u, 0u, 0u, 0u};
        for (int i = lane; i < (16 * HDIM) / 8; i += 32)
            __builtin_memcpy(hb + i * 8, &z, 16);
    }
    v8f c[8];
    #pragma unroll
    for (int jt = 0; jt < 8; ++jt)
        #pragma unroll
        for (int e = 0; e < 8; ++e) c[jt][e] = 0.0f;

    // ---------------- encoder: 15 steps ----------------
    #pragma unroll 1
    for (int t = 0; t < OBS; ++t) {
        v16h ax{};
        if (lane < 16) {
            float xv[4];
            __builtin_memcpy(xv, xg + ((size_t)(base + lane) * OBS + t) * 4, 16);
            ax[0] = (_Float16)xv[0]; ax[1] = (_Float16)xv[1];
            ax[2] = (_Float16)xv[2]; ax[3] = (_Float16)xv[3];
            ax[4] = (_Float16)1.0f;                 // bias lane (Wx col 4)
        }
        v16h hf[4];
        #pragma unroll
        for (int kf = 0; kf < 4; ++kf) hf[kf] = load_hfrag(hb, kf, n16, hi);
        // last step: capture h_enc into c (decoder starts with c0 = h_enc)
        lstm_step(WhhE, WxE, hb, ax, hf, c, n16, hi, t == OBS - 1);
    }

    // ---------------- decoder: 15 steps ----------------
    // x0 = x[:, -1, :]
    v16h ax{};
    if (lane < 16) {
        float xv[4];
        __builtin_memcpy(xv, xg + ((size_t)(base + lane) * OBS + (OBS - 1)) * 4, 16);
        ax[0] = (_Float16)xv[0]; ax[1] = (_Float16)xv[1];
        ax[2] = (_Float16)xv[2]; ax[3] = (_Float16)xv[3];
        ax[4] = (_Float16)1.0f;
    }
    // h fragments of h_enc (reused/refreshed each iteration)
    v16h hf[4];
    #pragma unroll
    for (int kf = 0; kf < 4; ++kf) hf[kf] = load_hfrag(hb, kf, n16, hi);

    #pragma unroll 1
    for (int t = 0; t < PRED; ++t) {
        lstm_step(WhhD, WxD, hb, ax, hf, c, n16, hi, false);

        // refresh fragments with h_new: used by out-projection AND next step
        #pragma unroll
        for (int kf = 0; kf < 4; ++kf) hf[kf] = load_hfrag(hb, kf, n16, hi);

        // out projection: x0 = h_new @ out_W^T + out_b   (N=4 of 16 used)
        v8f oa{};
        #pragma unroll
        for (int kf = 0; kf < 4; ++kf) {
            v16h bo{};
            if (n16 < 4) bo = ld32h(outW + (size_t)n16 * HDIM + kf * 32 + hi * 16);
            oa = __builtin_amdgcn_wmma_f32_16x16x32_f16(false, hf[kf], false, bo, (short)0, oa, false, false);
        }

        // write predictions + stage x0 for the next step (tiny LDS transpose)
        const float obv = obp[n16];
        if (n16 < 4) {
            #pragma unroll
            for (int r = 0; r < 8; ++r) {
                const int m = r + 8 * hi;
                const float v = oa[r] + obv;        // add out_b only on active lanes
                out[((size_t)(base + m) * PRED + t) * 4 + n16] = v;
                xb[m * 4 + n16] = (_Float16)v;
            }
        }
        v16h ax2{};
        if (lane < 16) {
            __builtin_memcpy(&ax2, xb + lane * 4, 8);  // 4 halves
            ax2[4] = (_Float16)1.0f;
        }
        ax = ax2;
    }
}

// ---------------------------------------------------------------------------
extern "C" void kernel_launch(void* const* d_in, const int* in_sizes, int n_in,
                              void* d_out, int out_size, void* d_ws, size_t ws_size,
                              hipStream_t stream) {
    const float* x      = (const float*)d_in[0];
    const float* eWih   = (const float*)d_in[1];
    const float* eWhh   = (const float*)d_in[2];
    const float* ebih   = (const float*)d_in[3];
    const float* ebhh   = (const float*)d_in[4];
    const float* dWih   = (const float*)d_in[5];
    const float* dWhh   = (const float*)d_in[6];
    const float* dbih   = (const float*)d_in[7];
    const float* dbhh   = (const float*)d_in[8];
    const float* oW     = (const float*)d_in[9];
    const float* ob     = (const float*)d_in[10];
    float* out          = (float*)d_out;
    unsigned char* ws   = (unsigned char*)d_ws;

    // pack weights -> f16 LDS image in workspace
    seqlstm_prep<<<256, 256, 0, stream>>>(eWih, eWhh, ebih, ebhh,
                                          dWih, dWhh, dbih, dbhh, oW, ob, ws);

    // 32768 samples / (16 per wave * 8 waves) = 256 blocks
    seqlstm_main<<<BATCH / (16 * WAVES), TPB, SMEM_BYTES, stream>>>(x, ws, out);
}
